// MultiHeadSelfAttention_84791244358145
// MI455X (gfx1250) — matrix operand order repaired and checked by reading, not compiled, not run on hardware
//
#include <hip/hip_runtime.h>

// ---------------------------------------------------------------------------
// MHA forward for MI455X (gfx1250): bf16 WMMA (16x16x32, f32 accum)
// B=4, S=2048, E=1024, H=16, D=64
// GEMMs: 64x64 per wave (4x4 WMMA tiles), single-wave workgroups so all
// tile-level control flow / epilogue bases are scalar (SGPR) values.
// ---------------------------------------------------------------------------

#define BATCH 4
#define SEQ   2048
#define EMB   1024
#define NHEAD 16
#define HDIM  64
#define MROWS (BATCH * SEQ)        // 8192
#define NQKV  (3 * EMB)            // 3072

typedef __attribute__((ext_vector_type(16))) __bf16 v16bf;
typedef __attribute__((ext_vector_type(8)))  __bf16 v8bf;
typedef __attribute__((ext_vector_type(8)))  float  v8f;

__device__ __forceinline__ v8f wmma_bf16(v16bf a, v16bf b, v8f c) {
  // (neg_a, A, neg_b, B, c_mod, C, reuse_a, reuse_b)
  return __builtin_amdgcn_wmma_f32_16x16x32_bf16(false, a, false, b, (short)0, c,
                                                 false, false);
}

__device__ __forceinline__ v16bf cat8(v8bf lo, v8bf hi) {
  v16bf r;
#pragma unroll
  for (int i = 0; i < 8; ++i) { r[i] = lo[i]; r[i + 8] = hi[i]; }
  return r;
}

// ---------------------------------------------------------------------------
// Stage 0: fp32 -> bf16 conversion
// ---------------------------------------------------------------------------
__global__ void cvt_bf16_kernel(const float* __restrict__ src,
                                __bf16* __restrict__ dst, int n) {
  int i = blockIdx.x * blockDim.x + threadIdx.x;
  if (i < n) dst[i] = (__bf16)src[i];
}

// ---------------------------------------------------------------------------
// Stage 1: QKV projection.  qkv = x @ Wqkv^T + b.
// Single-wave workgroup -> one 64x64 output tile (4x4 accumulators).
// K=1024 in steps of 32: 4 A-frags + 4 B-frags -> 16 WMMAs.
// blockIdx-derived tile => three/h/bh are SGPRs; scatter branch is scalar.
// ---------------------------------------------------------------------------
__global__ __launch_bounds__(32, 1)
void qkv_gemm_kernel(const __bf16* __restrict__ xb,
                     const __bf16* __restrict__ wq,   // [3072][1024]
                     const float*  __restrict__ bias, // [3072]
                     __bf16* __restrict__ qb,
                     __bf16* __restrict__ kb,
                     __bf16* __restrict__ vT) {
  const int lane = threadIdx.x;
  const int tile = blockIdx.x;
  const int NT = NQKV / 64;                 // 48 strip-tiles along N
  const int mBase = (tile / NT) * 64;       // uniform (SGPR)
  const int nBase = (tile % NT) * 64;       // uniform (SGPR)
  const int hi = lane >> 4;
  const int l16 = lane & 15;

  const __bf16* arow = xb + (size_t)(mBase + l16) * EMB + hi * 16;
  const __bf16* brow = wq + (size_t)(nBase + l16) * EMB + hi * 16;

  v8f acc[4][4];
#pragma unroll
  for (int i = 0; i < 4; ++i)
#pragma unroll
    for (int j = 0; j < 4; ++j)
      acc[i][j] = (v8f){0.f, 0.f, 0.f, 0.f, 0.f, 0.f, 0.f, 0.f};

  for (int k = 0; k < EMB; k += 32) {
    __builtin_prefetch(arow + k + 512, 0, 0);   // global_prefetch_b8
    __builtin_prefetch(brow + k + 512, 0, 0);
    v16bf a[4], b[4];
#pragma unroll
    for (int i = 0; i < 4; ++i) {
      a[i] = cat8(*(const v8bf*)(arow + k + i * (16 * EMB)),
                  *(const v8bf*)(arow + k + i * (16 * EMB) + 8));
      b[i] = *(const v16bf*)(brow + k + i * (16 * EMB));
    }
#pragma unroll
    for (int i = 0; i < 4; ++i)
#pragma unroll
      for (int j = 0; j < 4; ++j)
        acc[i][j] = wmma_bf16(a[i], b[j], acc[i][j]);
  }

  // Uniform epilogue parameters: the 64-wide strip is one (three, head) and
  // the 64-row strip is one batch (SEQ % 64 == 0).
  const int three = nBase >> 10;            // uniform
  const int h = (nBase >> 6) & 15;          // uniform
  const int bI = mBase >> 11;               // uniform
  const int sBase = mBase & (SEQ - 1);      // uniform
  const int bh = bI * NHEAD + h;            // uniform

  if (three != 2) {
    // q or k: [bh][s][d] row-major
    __bf16* dstBase = (three == 0 ? qb : kb) +
                      ((size_t)bh * SEQ + sBase) * HDIM;
#pragma unroll
    for (int j = 0; j < 4; ++j) {
      const float bv = bias[nBase + 16 * j + l16];
      const int d = 16 * j + l16;
#pragma unroll
      for (int i = 0; i < 4; ++i)
#pragma unroll
        for (int r = 0; r < 8; ++r)
          dstBase[(size_t)(16 * i + r + hi * 8) * HDIM + d] =
              (__bf16)(acc[i][j][r] + bv);
    }
  } else {
    // v: transposed [bh][d][s]
    __bf16* dstBase = vT + ((size_t)bh * HDIM) * SEQ + sBase;
#pragma unroll
    for (int j = 0; j < 4; ++j) {
      const float bv = bias[nBase + 16 * j + l16];
      const int d = 16 * j + l16;
#pragma unroll
      for (int i = 0; i < 4; ++i)
#pragma unroll
        for (int r = 0; r < 8; ++r)
          dstBase[(size_t)d * SEQ + 16 * i + r + hi * 8] =
              (__bf16)(acc[i][j][r] + bv);
    }
  }
}

// ---------------------------------------------------------------------------
// Stage 2: flash attention.  Block = 4 waves; wave = 32 query rows (2 tiles
// sharing each K/V fragment).  Key tiles of 32: 8 WMMAs scores + 8 WMMAs PV.
// ---------------------------------------------------------------------------
__global__ __launch_bounds__(128, 1)
void attention_kernel(const __bf16* __restrict__ qb,
                      const __bf16* __restrict__ kb,
                      const __bf16* __restrict__ vT,
                      __bf16* __restrict__ attn) {
  __shared__ __align__(16) __bf16 pbuf[4][2][16][32];

  const int lane = threadIdx.x & 31;
  const int wave = threadIdx.x >> 5;
  const int bh = blockIdx.x >> 4;            // 16 blocks per (b,h)
  const int qBlk = blockIdx.x & 15;
  const int bI = bh >> 4;
  const int h = bh & 15;
  const int qBase = qBlk * 128 + wave * 32;
  const int hi = lane >> 4;
  const int l16 = lane & 15;

  const __bf16* qp = qb + (size_t)bh * SEQ * HDIM;
  const __bf16* kp = kb + (size_t)bh * SEQ * HDIM + (size_t)l16 * HDIM + hi * 16;
  const __bf16* vp = vT + (size_t)bh * HDIM * SEQ + (size_t)l16 * SEQ + hi * 16;

  // Preload q A-fragments: 2 row-tiles x 2 d-chunks ([0,32) and [32,64))
  v16bf aq[2][2];
#pragma unroll
  for (int rt = 0; rt < 2; ++rt) {
    const __bf16* qrow = qp + (size_t)(qBase + 16 * rt + l16) * HDIM + hi * 16;
    aq[rt][0] = cat8(*(const v8bf*)(qrow),      *(const v8bf*)(qrow + 8));
    aq[rt][1] = cat8(*(const v8bf*)(qrow + 32), *(const v8bf*)(qrow + 40));
  }

  float mrow[2][8], lsum[2][8];
#pragma unroll
  for (int rt = 0; rt < 2; ++rt)
#pragma unroll
    for (int r = 0; r < 8; ++r) { mrow[rt][r] = -1e30f; lsum[rt][r] = 0.f; }
  v8f o[2][4];
#pragma unroll
  for (int rt = 0; rt < 2; ++rt)
#pragma unroll
    for (int dt = 0; dt < 4; ++dt)
      o[rt][dt] = (v8f){0.f, 0.f, 0.f, 0.f, 0.f, 0.f, 0.f, 0.f};
  const float scale = 0.125f;                // 1/sqrt(64)

  for (int t = 0; t < SEQ; t += 32) {
    // ---- K fragments shared by both row-tiles (const tile offsets) ----
    const __bf16* kt = kp + (size_t)t * HDIM;
    const v16bf kf0a = *(const v16bf*)(kt);
    const v16bf kf0b = *(const v16bf*)(kt + 32);
    const v16bf kf1a = *(const v16bf*)(kt + 16 * HDIM);
    const v16bf kf1b = *(const v16bf*)(kt + 16 * HDIM + 32);

#pragma unroll
    for (int rt = 0; rt < 2; ++rt) {
      v8f s0 = {0.f,0.f,0.f,0.f,0.f,0.f,0.f,0.f};
      v8f s1 = s0;
      s0 = wmma_bf16(aq[rt][0], kf0a, s0);
      s0 = wmma_bf16(aq[rt][1], kf0b, s0);
      s1 = wmma_bf16(aq[rt][0], kf1a, s1);
      s1 = wmma_bf16(aq[rt][1], kf1b, s1);
      // ---- online softmax (row = r + 8*hi; cols across 16-lane half) ----
#pragma unroll
      for (int r = 0; r < 8; ++r) {
        float x0 = s0[r] * scale;
        float x1 = s1[r] * scale;
        float mx = fmaxf(x0, x1);
        mx = fmaxf(mx, __shfl_xor(mx, 1, 32));
        mx = fmaxf(mx, __shfl_xor(mx, 2, 32));
        mx = fmaxf(mx, __shfl_xor(mx, 4, 32));
        mx = fmaxf(mx, __shfl_xor(mx, 8, 32));
        const float mnew = fmaxf(mrow[rt][r], mx);
        const float corr = __expf(mrow[rt][r] - mnew);
        const float p0 = __expf(x0 - mnew);
        const float p1 = __expf(x1 - mnew);
        float ps = p0 + p1;
        ps += __shfl_xor(ps, 1, 32);
        ps += __shfl_xor(ps, 2, 32);
        ps += __shfl_xor(ps, 4, 32);
        ps += __shfl_xor(ps, 8, 32);
        lsum[rt][r] = lsum[rt][r] * corr + ps;
        mrow[rt][r] = mnew;
        o[rt][0][r] *= corr; o[rt][1][r] *= corr;
        o[rt][2][r] *= corr; o[rt][3][r] *= corr;
        const int prow = r + hi * 8;
        pbuf[wave][rt][prow][l16]      = (__bf16)p0;
        pbuf[wave][rt][prow][l16 + 16] = (__bf16)p1;
      }
    }
    __syncthreads();
    // ---- reshape P: C-layout -> A-layout via LDS ----
    v16bf ap[2];
#pragma unroll
    for (int rt = 0; rt < 2; ++rt) {
      const __bf16* pr = &pbuf[wave][rt][l16][hi * 16];
      ap[rt] = cat8(*(const v8bf*)(pr), *(const v8bf*)(pr + 8));
    }
    // ---- attn += P(16x32) @ V(32x64); V fragments shared by row-tiles ----
#pragma unroll
    for (int dt = 0; dt < 4; ++dt) {
      const v16bf vf = *(const v16bf*)(vp + (size_t)dt * (16 * SEQ) + t);
      o[0][dt] = wmma_bf16(ap[0], vf, o[0][dt]);
      o[1][dt] = wmma_bf16(ap[1], vf, o[1][dt]);
    }
    __syncthreads();
  }

  // ---- epilogue: normalize and store attn[b][s][h*64+d] (bf16) ----
#pragma unroll
  for (int rt = 0; rt < 2; ++rt) {
#pragma unroll
    for (int r = 0; r < 8; ++r) {
      const float inv = 1.0f / lsum[rt][r];
      const int srow = qBase + 16 * rt + r + hi * 8;
      __bf16* dst = attn + ((size_t)(bI * SEQ + srow)) * EMB + h * HDIM + l16;
      dst[0]  = (__bf16)(o[rt][0][r] * inv);
      dst[16] = (__bf16)(o[rt][1][r] * inv);
      dst[32] = (__bf16)(o[rt][2][r] * inv);
      dst[48] = (__bf16)(o[rt][3][r] * inv);
    }
  }
}

// ---------------------------------------------------------------------------
// Stage 3: output projection.  out = attn @ Wfc^T + b_fc  (fp32 out)
// Single-wave workgroup, 64x64 tile, 4x4 accumulators.
// ---------------------------------------------------------------------------
__global__ __launch_bounds__(32, 1)
void out_gemm_kernel(const __bf16* __restrict__ ab,
                     const __bf16* __restrict__ wf,   // [1024][1024]
                     const float*  __restrict__ bias, // [1024]
                     float* __restrict__ out) {
  const int lane = threadIdx.x;
  const int tile = blockIdx.x;
  const int NT = EMB / 64;                  // 16
  const int mBase = (tile / NT) * 64;       // uniform
  const int nBase = (tile % NT) * 64;       // uniform
  const int hi = lane >> 4;
  const int l16 = lane & 15;

  const __bf16* arow = ab + (size_t)(mBase + l16) * EMB + hi * 16;
  const __bf16* brow = wf + (size_t)(nBase + l16) * EMB + hi * 16;

  v8f acc[4][4];
#pragma unroll
  for (int i = 0; i < 4; ++i)
#pragma unroll
    for (int j = 0; j < 4; ++j)
      acc[i][j] = (v8f){0.f, 0.f, 0.f, 0.f, 0.f, 0.f, 0.f, 0.f};

  for (int k = 0; k < EMB; k += 32) {
    __builtin_prefetch(arow + k + 512, 0, 0);
    __builtin_prefetch(brow + k + 512, 0, 0);
    v16bf a[4], b[4];
#pragma unroll
    for (int i = 0; i < 4; ++i) {
      a[i] = cat8(*(const v8bf*)(arow + k + i * (16 * EMB)),
                  *(const v8bf*)(arow + k + i * (16 * EMB) + 8));
      b[i] = *(const v16bf*)(brow + k + i * (16 * EMB));
    }
#pragma unroll
    for (int i = 0; i < 4; ++i)
#pragma unroll
      for (int j = 0; j < 4; ++j)
        acc[i][j] = wmma_bf16(a[i], b[j], acc[i][j]);
  }

  float* dstBase = out + (size_t)mBase * EMB + nBase;
#pragma unroll
  for (int j = 0; j < 4; ++j) {
    const float bv = bias[nBase + 16 * j + l16];
#pragma unroll
    for (int i = 0; i < 4; ++i)
#pragma unroll
      for (int r = 0; r < 8; ++r)
        dstBase[(size_t)(16 * i + r + hi * 8) * EMB + 16 * j + l16] =
            acc[i][j][r] + bv;
  }
}

// ---------------------------------------------------------------------------
// Launch
// ---------------------------------------------------------------------------
extern "C" void kernel_launch(void* const* d_in, const int* in_sizes, int n_in,
                              void* d_out, int out_size, void* d_ws, size_t ws_size,
                              hipStream_t stream) {
  (void)in_sizes; (void)n_in; (void)out_size; (void)ws_size;
  const float* x    = (const float*)d_in[0];   // [4,2048,1024]
  const float* Wqkv = (const float*)d_in[1];   // [3072,1024]
  const float* bqkv = (const float*)d_in[2];   // [3072]
  const float* Wfc  = (const float*)d_in[3];   // [1024,1024]
  const float* bfc  = (const float*)d_in[4];   // [1024]
  float* out = (float*)d_out;                  // [4,2048,1024]

  char* ws = (char*)d_ws;
  const size_t SZ_X    = (size_t)MROWS * EMB * 2;     // 16 MB
  const size_t SZ_WQKV = (size_t)NQKV * EMB * 2;      //  6 MB
  const size_t SZ_WFC  = (size_t)EMB * EMB * 2;       //  2 MB
  const size_t SZ_HEAD = (size_t)BATCH * NHEAD * SEQ * HDIM * 2; // 16 MB each
  __bf16* xb  = (__bf16*)(ws);
  __bf16* wqb = (__bf16*)(ws + SZ_X);
  __bf16* wfb = (__bf16*)(ws + SZ_X + SZ_WQKV);
  __bf16* qb  = (__bf16*)(ws + SZ_X + SZ_WQKV + SZ_WFC);
  __bf16* kb  = (__bf16*)(ws + SZ_X + SZ_WQKV + SZ_WFC + SZ_HEAD);
  __bf16* vT  = (__bf16*)(ws + SZ_X + SZ_WQKV + SZ_WFC + 2 * SZ_HEAD);
  __bf16* at  = (__bf16*)(ws + SZ_X + SZ_WQKV + SZ_WFC + 3 * SZ_HEAD);

  // Stage 0: conversions
  {
    int n;
    n = MROWS * EMB;
    cvt_bf16_kernel<<<(n + 255) / 256, 256, 0, stream>>>(x, xb, n);
    n = NQKV * EMB;
    cvt_bf16_kernel<<<(n + 255) / 256, 256, 0, stream>>>(Wqkv, wqb, n);
    n = EMB * EMB;
    cvt_bf16_kernel<<<(n + 255) / 256, 256, 0, stream>>>(Wfc, wfb, n);
  }
  // Stage 1: QKV GEMM — (8192/64)*(3072/64)=6144 single-wave workgroups
  qkv_gemm_kernel<<<6144, 32, 0, stream>>>(xb, wqb, bqkv, qb, kb, vT);
  // Stage 2: attention — 64 (b,h) * 16 blocks, 4 waves * 32 q-rows each
  attention_kernel<<<1024, 128, 0, stream>>>(qb, kb, vT, at);
  // Stage 3: output projection — (8192/64)*(1024/64)=2048 single-wave WGs
  out_gemm_kernel<<<2048, 32, 0, stream>>>(at, wfb, bfc, out);
}
